// QRNN_63565515981274
// MI455X (gfx1250) — matrix operand order repaired and checked by reading, not compiled, not run on hardware
//
#include <hip/hip_runtime.h>
#include <hip/hip_bf16.h>

// ---- problem constants --------------------------------------------------
#define BB    32
#define LL    4096
#define CIN   128
#define HH    256
#define C3H   768          // 3*H
#define KK    3
#define TT    4094         // L - K + 1
#define KSTEPS 12          // 384 / 32
#define CTILES 48          // 768 / 16
#define TGT    8           // t-tiles swept per wave (B kept in registers)
#define TCHUNK 256
#define NCHUNK 16

typedef __attribute__((ext_vector_type(16))) _Float16 v16h;
typedef __attribute__((ext_vector_type(8)))  _Float16 v8h;
typedef __attribute__((ext_vector_type(4)))  _Float16 v4h;
typedef __attribute__((ext_vector_type(8)))  float    v8f;
typedef __attribute__((ext_vector_type(4)))  float    v4f;

#define LOG2E 1.4426950408889634f

// hardware-native sigmoid/tanh: v_exp_f32 + v_rcp_f32 (TRANS ops that
// co-execute with XDL WMMA on CDNA5)
__device__ __forceinline__ float fast_sigmoid(float x) {
    return __builtin_amdgcn_rcpf(1.0f + __builtin_amdgcn_exp2f(-LOG2E * x));
}
__device__ __forceinline__ float fast_tanh(float x) {
    return 2.0f * __builtin_amdgcn_rcpf(1.0f + __builtin_amdgcn_exp2f(-2.0f * LOG2E * x)) - 1.0f;
}

// ---- kernel 1: convert x f32 -> f16 (once) ------------------------------
__global__ __launch_bounds__(256) void cvt_x_kernel(const float* __restrict__ x,
                                                    _Float16* __restrict__ xh) {
    size_t idx = (size_t)blockIdx.x * blockDim.x + threadIdx.x;
    v4f a = ((const v4f*)x)[idx];
    v4h r;
    r.x = (_Float16)a.x; r.y = (_Float16)a.y;
    r.z = (_Float16)a.z; r.w = (_Float16)a.w;
    ((v4h*)xh)[idx] = r;
}

// ---- kernel 2: pack W into B-matrix per-lane register layout ------------
// Bp[((s*48 + ct)*32 + lane)*16 + j]:
//   lanes 0-15 : column n = lane,    K = j        (j = 0..15)
//   lanes16-31 : column n = lane-16, K = 16 + j
// reduction index r = s*32 + K ;  k = r/128 ; i = r%128 ;  element W[c,i,k]
__global__ __launch_bounds__(256) void pack_w_kernel(const float* __restrict__ W,
                                                     _Float16* __restrict__ Bp) {
    int idx = blockIdx.x * 256 + threadIdx.x;        // 12*48*32*16 = 294912
    if (idx >= KSTEPS * CTILES * 32 * 16) return;
    int j     = idx & 15;
    int lane  = (idx >> 4) & 31;
    int sc    = idx >> 9;                             // s*48 + ct
    int ctile = sc % CTILES;
    int s     = sc / CTILES;
    int n  = lane & 15;
    int kl = (lane < 16) ? j : (16 + j);
    int r  = s * 32 + kl;
    int k  = r >> 7;            // / 128
    int i  = r & 127;
    int c  = ctile * 16 + n;
    Bp[idx] = (_Float16)W[(c * CIN + i) * KK + k];
}

// ---- kernel 3: conv-as-GEMM via WMMA ------------------------------------
// One wave owns one c-tile: loads its 12 B fragments into registers ONCE,
// then sweeps TGT=8 t-tiles (one wmma per k-step per tile -> 2 A-loads/wmma,
// zero steady-state B traffic). Fused bias + hw-native activation epilogue.
// gates layout: gates[((tl*3 + g)*B + b)*H + h]   (tl local to chunk)
__global__ __launch_bounds__(256) void conv_wmma_kernel(const _Float16* __restrict__ xh,
                                                        const _Float16* __restrict__ Bp,
                                                        const float* __restrict__ bias,
                                                        float* __restrict__ gates,
                                                        int t0) {
    const int wid  = blockIdx.x * 8 + (threadIdx.x >> 5);   // 3072 waves/chunk
    const int lane = threadIdx.x & 31;
    const int ctile  = wid % CTILES;          // 0..47
    const int tgroup = wid / CTILES;          // 0..63
    const int tm   = lane & 15;               // A-matrix row M for this lane
    const int hi   = lane >> 4;               // half-wave select
    const int Nn   = lane & 15;               // C/D column per lane

    // ---- B fragments: register-resident for the whole wave --------------
    v16h Bfrag[KSTEPS];
#pragma unroll
    for (int s = 0; s < KSTEPS; ++s) {
        Bfrag[s] = *(const v16h*)(Bp + (((size_t)(s * CTILES + ctile) * 32 + lane) << 4));
    }

    // per-wave uniform epilogue constants
    const int c   = ctile * 16 + Nn;          // global output channel
    const float bv = bias[c];
    const int g   = ctile >> 4;               // 0=Z, 1=F, 2=O (uniform per wave)
    const int hh  = c & 255;
    const int mbase = hi * 8;                 // C/D: VGPR v -> M = v (+8 hi half)

#pragma unroll 1
    for (int tt = 0; tt < TGT; ++tt) {
        const int ntile = tgroup * TGT + tt;  // 0..511
        const int b     = ntile >> 4;
        const int ttile = (ntile & 15) << 4;  // local t of tile row 0

        v8f acc = {};
#pragma unroll
        for (int s = 0; s < KSTEPS; ++s) {
            const int k  = s >> 2;            // r = s*32 + kl ; k = r/128
            const int ib = (s & 3) << 5;      // i base within channel dim
            int trow = t0 + ttile + tm + k;   // x row (clamped; garbage rows
            if (trow > LL - 1) trow = LL - 1; //  masked at the store)
            const _Float16* arow = xh + (((size_t)b * LL + trow) << 7);  // *CIN
            // A k order used here (same as Bp): lanes0-15 hold K 0-15; lanes16-31 K 16-31
            const int i0 = ib + (hi ? 16 : 0);
            v8h alo = *(const v8h*)(arow + i0);
            v8h ahi = *(const v8h*)(arow + i0 + 8);
            v16h A = __builtin_shufflevector(alo, ahi,
                      0,1,2,3,4,5,6,7, 8,9,10,11,12,13,14,15);
            acc = __builtin_amdgcn_wmma_f32_16x16x32_f16(false, A, false, Bfrag[s],
                                                         (short)0, acc, false, false);
        }

        // ---- epilogue: bias + activation + scatter to gates -------------
#pragma unroll
        for (int v = 0; v < 8; ++v) {
            const int tl = ttile + mbase + v; // local t of this element
            if (t0 + tl < TT) {
                float val = acc[v] + bv;
                val = (g == 0) ? fast_tanh(val) : fast_sigmoid(val);
                gates[((size_t)(tl * 3 + g) * BB + b) * HH + hh] = val;
            }
        }
    }
}

// ---- kernel 4: sequential fo-pool over one chunk ------------------------
__global__ __launch_bounds__(256) void scan_kernel(const float* __restrict__ gates,
                                                   float* __restrict__ hstate,
                                                   float* __restrict__ out,
                                                   int tc, int isfirst, int islast) {
    const int idx = blockIdx.x * 256 + threadIdx.x;   // 0..8191  (b*H + h)
    float h = isfirst ? 0.0f : hstate[idx];
    const float* gz = gates + idx;                    // +g*8192 selects gate
    int tl = 0;
#pragma unroll 4
    for (; tl + 4 <= tc; tl += 4) {
#pragma unroll
        for (int u = 0; u < 4; ++u) {
            const float* p = gz + (size_t)(tl + u) * 3 * (BB * HH);
            float z = p[0];
            float f = p[BB * HH];
            float o = p[2 * BB * HH];
            h = o * __builtin_fmaf(f, h, (1.0f - f) * z);
        }
    }
    for (; tl < tc; ++tl) {
        const float* p = gz + (size_t)tl * 3 * (BB * HH);
        float z = p[0];
        float f = p[BB * HH];
        float o = p[2 * BB * HH];
        h = o * __builtin_fmaf(f, h, (1.0f - f) * z);
    }
    hstate[idx] = h;
    if (islast) out[idx] = h;
}

// ---- host-side orchestration -------------------------------------------
extern "C" void kernel_launch(void* const* d_in, const int* in_sizes, int n_in,
                              void* d_out, int out_size, void* d_ws, size_t ws_size,
                              hipStream_t stream) {
    const float* x    = (const float*)d_in[0];   // (B, L, CIN) f32
    const float* W    = (const float*)d_in[1];   // (3H, CIN, K) f32
    const float* bias = (const float*)d_in[2];   // (3H,) f32
    float* out        = (float*)d_out;           // (B, H) f32

    // workspace carve-up (all offsets 256B aligned)
    char* ws = (char*)d_ws;
    _Float16* xh     = (_Float16*)ws;                                 // 33,554,432 B
    _Float16* Bp     = (_Float16*)(ws + 33554432);                    //    589,824 B
    float*    gates  = (float*)(ws + 33554432 + 589824);              // 25,165,824 B
    float*    hstate = (float*)(ws + 33554432 + 589824 + 25165824);   //     32,768 B

    // 1) x -> f16 (16,777,216 elements, 4/thread)
    cvt_x_kernel<<<(BB * LL * CIN) / (256 * 4), 256, 0, stream>>>(x, xh);

    // 2) pack weights into WMMA B-fragment layout
    pack_w_kernel<<<(KSTEPS * CTILES * 32 * 16 + 255) / 256, 256, 0, stream>>>(W, Bp);

    // 3) alternate: WMMA conv chunk -> sequential scan chunk
    for (int cidx = 0; cidx < NCHUNK; ++cidx) {
        const int t0 = cidx * TCHUNK;
        // waves = 48 c-tiles * 64 t-groups = 3072 -> 384 blocks of 8 waves
        conv_wmma_kernel<<<(CTILES * (512 / TGT)) / 8, 256, 0, stream>>>(xh, Bp, bias,
                                                                         gates, t0);
        int tc = TT - t0; if (tc > TCHUNK) tc = TCHUNK;
        scan_kernel<<<(BB * HH) / 256, 256, 0, stream>>>(gates, hstate, out, tc,
                                                         cidx == 0, cidx == NCHUNK - 1);
    }
}
